// RiskGNN_14508399526529
// MI455X (gfx1250) — compile-verified, matrix operand order checked by reading, not run on hardware
//
#include <hip/hip_runtime.h>

// ---------------------------------------------------------------------------
// RiskGNN forward for MI455X (gfx1250): bf16 WMMA GEMMs + relation-bucketed
// edge matmul (permuted layout) + atomic segment softmax. wave32.
// ---------------------------------------------------------------------------

typedef __bf16 bf16;
typedef __attribute__((ext_vector_type(16))) __bf16 v16bf;
typedef __attribute__((ext_vector_type(8)))  __bf16 v8bf;
typedef __attribute__((ext_vector_type(8)))  float  v8f;

constexpr int DD   = 128;   // hidden dim
constexpr int WSTR = 136;   // padded LDS stride (bf16) -> conflict-free ds_load_b128
constexpr int CPT  = 4;     // channels per thread in edge streaming kernels

// ---- ordered-uint encoding so atomicMax(uint) implements float max --------
__device__ __forceinline__ unsigned fkey(float f) {
  unsigned u = __float_as_uint(f);
  return (u & 0x80000000u) ? ~u : (u | 0x80000000u);
}
__device__ __forceinline__ float fdec(unsigned k) {
  return (k & 0x80000000u) ? __uint_as_float(k & 0x7FFFFFFFu)
                           : __uint_as_float(~k);
}

// ---- WMMA fragment loaders (ISA 7.12.2 16-bit layouts, wave32) ------------
__device__ __forceinline__ v16bf zero_frag() {
  v16bf a;
#pragma unroll
  for (int i = 0; i < 16; ++i) a[i] = (bf16)0.0f;
  return a;
}
// A 16x32 bf16: lanes 0-15 row M=lane&15, K=kb+0..7 (v0..3), kb+16..23 (v4..7);
// lanes 16-31 same rows, kb += 8.
__device__ __forceinline__ v16bf load_a_f32(const float* rowp, int k0) {
  int lane = threadIdx.x & 31;
  int kb = k0 + ((lane & 16) ? 8 : 0);
  float4 w0 = *(const float4*)(rowp + kb);
  float4 w1 = *(const float4*)(rowp + kb + 4);
  float4 w2 = *(const float4*)(rowp + kb + 16);
  float4 w3 = *(const float4*)(rowp + kb + 20);
  v16bf a;
  a[0]=(bf16)w0.x;  a[1]=(bf16)w0.y;  a[2]=(bf16)w0.z;  a[3]=(bf16)w0.w;
  a[4]=(bf16)w1.x;  a[5]=(bf16)w1.y;  a[6]=(bf16)w1.z;  a[7]=(bf16)w1.w;
  a[8]=(bf16)w2.x;  a[9]=(bf16)w2.y;  a[10]=(bf16)w2.z; a[11]=(bf16)w2.w;
  a[12]=(bf16)w3.x; a[13]=(bf16)w3.y; a[14]=(bf16)w3.z; a[15]=(bf16)w3.w;
  return a;
}
__device__ __forceinline__ v16bf load_a_bf16(const bf16* rowp, int k0) {
  int lane = threadIdx.x & 31;
  int kb = k0 + ((lane & 16) ? 8 : 0);
  v8bf lo = *(const v8bf*)(rowp + kb);        // 16B -> global_load_b128
  v8bf hi = *(const v8bf*)(rowp + kb + 16);
  v16bf a;
#pragma unroll
  for (int i = 0; i < 8; ++i) { a[i] = lo[i]; a[8 + i] = hi[i]; }
  return a;
}
// B 32x16 bf16 from LDS W^T[c][k] (stride WSTR): lanes 0-15 col N=lane,
// K=kb+0..7 / kb+16..23; lanes 16-31 kb += 8.
__device__ __forceinline__ v16bf load_b_lds(const bf16* Wl, int k0, int n0) {
  int lane = threadIdx.x & 31;
  int c  = n0 + (lane & 15);
  int kb = k0 + ((lane & 16) ? 8 : 0);
  const bf16* p = Wl + c * WSTR + kb;
  v8bf lo = *(const v8bf*)(p);                // ds_load_b128
  v8bf hi = *(const v8bf*)(p + 16);
  v16bf b;
#pragma unroll
  for (int i = 0; i < 8; ++i) { b[i] = lo[i]; b[8 + i] = hi[i]; }
  return b;
}
__device__ __forceinline__ void stage_w_lds(bf16* Wl, const float* W, int t) {
  // W row-major [k][c] f32 -> LDS transposed bf16 W^T[c][k], stride WSTR
  for (int i = t; i < DD * DD / 4; i += 256) {
    float4 w = ((const float4*)W)[i];
    int k = (i * 4) >> 7;
    int c = (i * 4) & (DD - 1);
    Wl[(c + 0) * WSTR + k] = (bf16)w.x;
    Wl[(c + 1) * WSTR + k] = (bf16)w.y;
    Wl[(c + 2) * WSTR + k] = (bf16)w.z;
    Wl[(c + 3) * WSTR + k] = (bf16)w.w;
  }
}

// ---------------------------------------------------------------------------
// GEMM: C[M,128] = A[M,128] @ W[128,128] + bias.  8 waves x 16 rows / block.
// ---------------------------------------------------------------------------
template <bool WF, bool WB>
__global__ __launch_bounds__(256) void gemm_node_kernel(
    const float* __restrict__ A, const float* __restrict__ W,
    const float* __restrict__ bias, float* __restrict__ Cf,
    bf16* __restrict__ Cb, int M)
{
  __shared__ bf16 Wl[DD * WSTR];
  int t = threadIdx.x;
  stage_w_lds(Wl, W, t);
  __syncthreads();

  int wave = t >> 5, lane = t & 31;
  int m0 = blockIdx.x * 128 + wave * 16;
  int cb = lane & 15;
  int mh = (lane & 16) ? 8 : 0;

  v16bf a[4];
  int arow = m0 + cb;
  if (arow < M) {
    const float* rowp = A + (size_t)arow * DD;
#pragma unroll
    for (int kt = 0; kt < 4; ++kt) a[kt] = load_a_f32(rowp, kt * 32);
  } else {
#pragma unroll
    for (int kt = 0; kt < 4; ++kt) a[kt] = zero_frag();
  }
  float bv[8];
#pragma unroll
  for (int nt = 0; nt < 8; ++nt) bv[nt] = bias[nt * 16 + cb];

  bool full = (m0 + 16) <= M;           // wave-uniform
  for (int nt = 0; nt < 8; ++nt) {
    v8f acc = {0.f, 0.f, 0.f, 0.f, 0.f, 0.f, 0.f, 0.f};
#pragma unroll
    for (int kt = 0; kt < 4; ++kt) {
      v16bf b = load_b_lds(Wl, kt * 32, nt * 16);
      acc = __builtin_amdgcn_wmma_f32_16x16x32_bf16(false, a[kt], false, b,
                                                    (short)0, acc, false, false);
    }
    int c = nt * 16 + cb;
    if (full) {
#pragma unroll
      for (int i = 0; i < 8; ++i) {
        size_t o = (size_t)(m0 + mh + i) * DD + c;
        float v = acc[i] + bv[nt];
        if constexpr (WF) Cf[o] = v;
        if constexpr (WB) Cb[o] = (bf16)v;
      }
    } else {
#pragma unroll
      for (int i = 0; i < 8; ++i) {
        int r = m0 + mh + i;
        if (r < M) {
          size_t o = (size_t)r * DD + c;
          float v = acc[i] + bv[nt];
          if constexpr (WF) Cf[o] = v;
          if constexpr (WB) Cb[o] = (bf16)v;
        }
      }
    }
  }
}

// ---------------------------------------------------------------------------
// Relation GEMM over bucketed edges, res stored in PERMUTED order:
// res[p] = x_bf16[pcol[p]] @ W_rel[r]  for p in [offs[r], offs[r+1]).
// Block = 128 permuted edges of one relation (8 waves x 16-edge tiles).
// ---------------------------------------------------------------------------
__global__ __launch_bounds__(256) void rel_gemm_kernel(
    const bf16* __restrict__ xbf, const float* __restrict__ Wrel,
    const int* __restrict__ offsets, const int* __restrict__ pcol,
    bf16* __restrict__ res)
{
  int b = blockIdx.x;
  int rel = -1, mblock = 0, base = 0;
#pragma unroll
  for (int r = 0; r < 8; ++r) {
    int cnt = offsets[r + 1] - offsets[r];
    int nb  = (cnt + 127) >> 7;
    if (rel < 0 && b < base + nb) { rel = r; mblock = b - base; }
    base += nb;
  }
  if (rel < 0) return;

  __shared__ bf16 Wl[DD * WSTR];
  int t = threadIdx.x;
  stage_w_lds(Wl, Wrel + (size_t)rel * DD * DD, t);
  __syncthreads();

  int wave = t >> 5, lane = t & 31;
  int e0   = offsets[rel] + mblock * 128 + wave * 16;
  int eEnd = offsets[rel + 1];
  if (e0 >= eEnd) return;

  int cb = lane & 15;
  int mh = (lane & 16) ? 8 : 0;
  bool full = (e0 + 16) <= eEnd;        // wave-uniform

  v16bf a[4];
  int ep = e0 + cb;
  if (ep < eEnd) {
    const bf16* rowp = xbf + (size_t)pcol[ep] * DD;
#pragma unroll
    for (int kt = 0; kt < 4; ++kt) a[kt] = load_a_bf16(rowp, kt * 32);
  } else {
#pragma unroll
    for (int kt = 0; kt < 4; ++kt) a[kt] = zero_frag();
  }

  for (int nt = 0; nt < 8; ++nt) {
    v8f acc = {0.f, 0.f, 0.f, 0.f, 0.f, 0.f, 0.f, 0.f};
#pragma unroll
    for (int kt = 0; kt < 4; ++kt) {
      v16bf bfr = load_b_lds(Wl, kt * 32, nt * 16);
      acc = __builtin_amdgcn_wmma_f32_16x16x32_bf16(false, a[kt], false, bfr,
                                                    (short)0, acc, false, false);
    }
    int c = nt * 16 + cb;
    if (full) {
#pragma unroll
      for (int i = 0; i < 8; ++i)
        res[(size_t)(e0 + mh + i) * DD + c] = (bf16)acc[i];
    } else {
#pragma unroll
      for (int i = 0; i < 8; ++i) {
        int p = e0 + mh + i;
        if (p < eEnd) res[(size_t)p * DD + c] = (bf16)acc[i];
      }
    }
  }
}

// ---------------------------------------------------------------------------
// small elementwise / graph kernels
// ---------------------------------------------------------------------------
__global__ void deg_kernel(const int* __restrict__ coli, float* deg, int E) {
  int i = blockIdx.x * 256 + threadIdx.x;
  if (i < E) atomicAdd(&deg[coli[i]], 1.0f);
}
__global__ void dis_kernel(const float* deg, float* dis, int N) {
  int i = blockIdx.x * 256 + threadIdx.x;
  if (i < N) { float d = deg[i]; dis[i] = d > 0.f ? rsqrtf(d) : 0.f; }
}
__global__ __launch_bounds__(256) void count_kernel(const int* et, int* counts, int E) {
  __shared__ int lc[8];
  int t = threadIdx.x;
  if (t < 8) lc[t] = 0;
  __syncthreads();
  int i = blockIdx.x * 256 + t;
  if (i < E) atomicAdd(&lc[et[i]], 1);
  __syncthreads();
  if (t < 8) atomicAdd(&counts[t], lc[t]);
}
__global__ void scan_kernel(const int* counts, int* offsets, int* cursor) {
  offsets[0] = 0;
  for (int r = 0; r < 8; ++r) {
    offsets[r + 1] = offsets[r] + counts[r];
    cursor[r] = offsets[r];
  }
}
// bucket edges: write permuted row/col index arrays (no perm indirection later)
__global__ void scatter_kernel(const int* __restrict__ et,
                               const int* __restrict__ rowi,
                               const int* __restrict__ coli,
                               int* cursor, int* prow, int* pcol, int E) {
  int i = blockIdx.x * 256 + threadIdx.x;
  if (i < E) {
    int p = atomicAdd(&cursor[et[i]], 1);
    prow[p] = rowi[i];
    pcol[p] = coli[i];
  }
}
__global__ __launch_bounds__(256) void bn_stats_kernel(
    const float* __restrict__ h, float* sums, long long total) {
  __shared__ float ls[DD], lq[DD];
  int t = threadIdx.x;
  if (t < DD) { ls[t] = 0.f; lq[t] = 0.f; }
  __syncthreads();
  long long i = (long long)blockIdx.x * 256 + t;
  long long stride = (long long)gridDim.x * 256;
  for (; i < total; i += stride) {
    float v = h[i]; int c = (int)(i & (DD - 1));
    atomicAdd(&ls[c], v); atomicAdd(&lq[c], v * v);
  }
  __syncthreads();
  if (t < DD) { atomicAdd(&sums[t], ls[t]); atomicAdd(&sums[DD + t], lq[t]); }
}
__global__ void bn_apply_kernel(const float* __restrict__ h, const float* sums,
                                const float* gamma, const float* beta,
                                float* cr, int N) {
  long long i = (long long)blockIdx.x * 256 + threadIdx.x;
  if (i < (long long)N * DD) {
    int c = (int)(i & (DD - 1));
    float mu  = sums[c] / (float)N;
    float var = sums[DD + c] / (float)N - mu * mu;
    float v = (h[i] - mu) * rsqrtf(var + 1e-5f) * gamma[c] + beta[c];
    cr[i] = v > 0.f ? v : 0.f;
  }
}
// seg-max over permuted res; 4 channels per thread
__global__ void edge_max_kernel(const bf16* __restrict__ res,
                                const int* __restrict__ prow,
                                unsigned* segkey, int E) {
  long long i = (long long)blockIdx.x * 256 + threadIdx.x;
  if (i < (long long)E * (DD / CPT)) {
    int e  = (int)(i >> 5);           // DD/CPT == 32
    int c0 = ((int)i & 31) * CPT;
    int r  = prow[e];
    const bf16* p = res + (size_t)e * DD + c0;
    unsigned* sk = segkey + (size_t)r * DD + c0;
#pragma unroll
    for (int j = 0; j < CPT; ++j) atomicMax(&sk[j], fkey((float)p[j]));
  }
}
// fused: softmax denominator/numerator + gcn message; 4 channels per thread
__global__ void edge_accum_kernel(const bf16* __restrict__ res,
                                  const bf16* __restrict__ xbf,
                                  const int* __restrict__ prow,
                                  const int* __restrict__ pcol,
                                  const float* __restrict__ dis,
                                  const unsigned* __restrict__ segkey,
                                  float* denom, float* numer, float* msggcn,
                                  int E) {
  long long i = (long long)blockIdx.x * 256 + threadIdx.x;
  if (i < (long long)E * (DD / CPT)) {
    __builtin_prefetch(res + (size_t)(i * CPT) + 16384, 0, 1);
    int e  = (int)(i >> 5);
    int c0 = ((int)i & 31) * CPT;
    int r = prow[e], s = pcol[e];
    float nw = dis[r] * dis[s];
    const bf16* rp = res + (size_t)e * DD + c0;
    const bf16* xp = xbf + (size_t)s * DD + c0;
    size_t o = (size_t)r * DD + c0;
#pragma unroll
    for (int j = 0; j < CPT; ++j) {
      float v  = (float)rp[j];
      float ex = __expf(v - fdec(segkey[o + j]));
      atomicAdd(&denom[o + j], ex);
      atomicAdd(&numer[o + j], v * ex);
      atomicAdd(&msggcn[o + j], (float)xp[j] * nw);
    }
  }
}
__global__ void combine_kernel(const float* numer, const float* denom,
                               const float* msggcn, float* comb, long long total) {
  long long i = (long long)blockIdx.x * 256 + threadIdx.x;
  if (i < total) {
    float msg = numer[i] / (denom[i] + 1e-16f);
    comb[i] = msggcn[i] + 0.5f * (msg > 0.f ? msg : 0.f);
  }
}
__global__ void gather_gelu_kernel(const float* __restrict__ cr,
                                   const int* __restrict__ idx,
                                   float* out, long long total) {
  long long i = (long long)blockIdx.x * 256 + threadIdx.x;
  if (i < total) {
    int n = idx[i >> 7];
    float v = cr[(size_t)n * DD + (i & (DD - 1))];
    out[i] = 0.5f * v * (1.0f + erff(v * 0.70710678118654752f));
  }
}

// ---------------------------------------------------------------------------
extern "C" void kernel_launch(void* const* d_in, const int* in_sizes, int n_in,
                              void* d_out, int out_size, void* d_ws, size_t ws_size,
                              hipStream_t stream) {
  const float* x        = (const float*)d_in[0];
  const int*   eidx     = (const int*)  d_in[1];
  const int*   idx      = (const int*)  d_in[2];
  const int*   etype    = (const int*)  d_in[3];
  const float* proj_w   = (const float*)d_in[5];
  const float* proj_b   = (const float*)d_in[6];
  const float* gamma    = (const float*)d_in[7];
  const float* beta     = (const float*)d_in[8];
  const float* Wrel[2]  = {(const float*)d_in[9],  (const float*)d_in[14]};
  const float* inw[2]   = {(const float*)d_in[10], (const float*)d_in[15]};
  const float* inb[2]   = {(const float*)d_in[11], (const float*)d_in[16]};
  const float* outw[2]  = {(const float*)d_in[12], (const float*)d_in[17]};
  const float* outb[2]  = {(const float*)d_in[13], (const float*)d_in[18]};

  const int N    = in_sizes[0] / DD;
  const int E    = in_sizes[3];
  const int NSEL = in_sizes[2];
  const int* rowi = eidx;
  const int* coli = eidx + E;
  const size_t ND = (size_t)N * DD;

  // ---- carve workspace ----
  char* base = (char*)d_ws; size_t cur = 0;
  auto carve = [&](size_t bytes) -> char* {
    char* p = base + cur; cur += (bytes + 255) & ~(size_t)255; return p;
  };
  float*    h      = (float*)   carve(ND * 4);
  float*    cr     = (float*)   carve(ND * 4);
  bf16*     xbf    = (bf16*)    carve(ND * 2);
  bf16*     res    = (bf16*)    carve((size_t)E * DD * 2);
  unsigned* segkey = (unsigned*)carve(ND * 4);
  float*    denom  = (float*)   carve(ND * 4);
  float*    numer  = (float*)   carve(ND * 4);
  float*    msggcn = (float*)   carve(ND * 4);
  float*    comb   = (float*)   carve(ND * 4);
  float*    dis    = (float*)   carve((size_t)N * 4);
  float*    deg    = (float*)   carve((size_t)N * 4);
  float*    sums   = (float*)   carve(2 * DD * 4);
  int*      counts = (int*)     carve(8 * 4);
  int*      offs   = (int*)     carve(9 * 4);
  int*      cursor = (int*)     carve(8 * 4);
  int*      prow   = (int*)     carve((size_t)E * 4);
  int*      pcol   = (int*)     carve((size_t)E * 4);
  (void)ws_size; (void)n_in; (void)out_size;

  const int TB = 256;
  const long long totEC = (long long)E * (DD / CPT);
  const int gE   = (E + TB - 1) / TB;
  const int gN   = (N + TB - 1) / TB;
  const int gND  = (int)((ND + TB - 1) / TB);
  const int gEC  = (int)((totEC + TB - 1) / TB);
  const int gblk = (N + 127) / 128;
  const int relblocks = (E + 127) / 128 + 8;

  // ---- graph prep: degree + relation buckets ----
  hipMemsetAsync(deg, 0, (size_t)N * 4, stream);
  hipMemsetAsync(counts, 0, 8 * 4, stream);
  hipMemsetAsync(sums, 0, 2 * DD * 4, stream);
  deg_kernel<<<gE, TB, 0, stream>>>(coli, deg, E);
  dis_kernel<<<gN, TB, 0, stream>>>(deg, dis, N);
  count_kernel<<<gE, TB, 0, stream>>>(etype, counts, E);
  scan_kernel<<<1, 1, 0, stream>>>(counts, offs, cursor);
  scatter_kernel<<<gE, TB, 0, stream>>>(etype, rowi, coli, cursor, prow, pcol, E);

  // ---- proj + BN + relu ----
  gemm_node_kernel<true, false><<<gblk, 256, 0, stream>>>(x, proj_w, proj_b,
                                                          h, nullptr, N);
  bn_stats_kernel<<<1024, 256, 0, stream>>>(h, sums, (long long)ND);
  bn_apply_kernel<<<gND, TB, 0, stream>>>(h, sums, gamma, beta, cr, N);

  // ---- two DAN layers ----
  for (int l = 0; l < 2; ++l) {
    gemm_node_kernel<false, true><<<gblk, 256, 0, stream>>>(cr, inw[l], inb[l],
                                                            nullptr, xbf, N);
    rel_gemm_kernel<<<relblocks, 256, 0, stream>>>(xbf, Wrel[l], offs, pcol, res);
    hipMemsetAsync(segkey, 0, ND * 4, stream);   // key 0 == below any real float
    hipMemsetAsync(denom,  0, ND * 4, stream);
    hipMemsetAsync(numer,  0, ND * 4, stream);
    hipMemsetAsync(msggcn, 0, ND * 4, stream);
    edge_max_kernel<<<gEC, TB, 0, stream>>>(res, prow, segkey, E);
    edge_accum_kernel<<<gEC, TB, 0, stream>>>(res, xbf, prow, pcol, dis, segkey,
                                              denom, numer, msggcn, E);
    combine_kernel<<<gND, TB, 0, stream>>>(numer, denom, msggcn, comb, (long long)ND);
    gemm_node_kernel<true, false><<<gblk, 256, 0, stream>>>(comb, outw[l], outb[l],
                                                            cr, nullptr, N);
  }

  // ---- gelu + gather ----
  const long long totO = (long long)NSEL * DD;
  gather_gelu_kernel<<<(int)((totO + TB - 1) / TB), TB, 0, stream>>>(
      cr, idx, (float*)d_out, totO);
}
